// PointnetFPModule_47571057771113
// MI455X (gfx1250) — matrix operand order repaired and checked by reading, not compiled, not run on hardware
//
#include <hip/hip_runtime.h>
#include <hip/hip_bf16.h>
#include <stdint.h>

#define BATCH 2
#define N1    16384
#define N2    4096
#define C1    128
#define C2    256
#define CIN0  384
#define COUT  256
#define BN_EPS 1e-5f

typedef __bf16 bf16_t;
typedef __bf16 v8bf  __attribute__((ext_vector_type(8)));
typedef __bf16 v16bf __attribute__((ext_vector_type(16)));
typedef float  v8f   __attribute__((ext_vector_type(8)));

// Async global->LDS path (ASYNCcnt) if the toolchain exposes the gfx1250
// builtins; falls back to VGPR-staged copies otherwise.  The builtin expects
// int4 pointers qualified with the global (AS1) / LDS (AS3) address spaces.
#if defined(__has_builtin)
#if __has_builtin(__builtin_amdgcn_global_load_async_to_lds_b128) && \
    __has_builtin(__builtin_amdgcn_s_wait_asynccnt)
#define USE_ASYNC_LDS 1
typedef int v4i_a __attribute__((ext_vector_type(4)));
typedef __attribute__((address_space(1))) v4i_a gas_v4i;
typedef __attribute__((address_space(3))) v4i_a las_v4i;
#endif
#endif

// ---------------------------------------------------------------- f32 -> bf16
__global__ __launch_bounds__(256) void f32_to_bf16_kernel(
    const float* __restrict__ s, bf16_t* __restrict__ d, int n) {
  int i = blockIdx.x * 256 + threadIdx.x;
  int stride = gridDim.x * 256;
  for (; i < n; i += stride) d[i] = (bf16_t)s[i];
}

// ---------------------------------------------------------------- 3-NN search
// sparse cloud staged in LDS (48KB), one thread per dense point, top-3 in regs
__global__ __launch_bounds__(256) void knn_kernel(
    const float* __restrict__ dxyz, const float* __restrict__ sxyz,
    int* __restrict__ idx, float* __restrict__ wgt) {
  __shared__ float sx[N2], sy[N2], sz[N2];
  const int b = blockIdx.y;
  const int tid = threadIdx.x;
  const float* S = sxyz + (size_t)b * 3 * N2;
  for (int j = tid; j < N2; j += 256) {
    sx[j] = S[j]; sy[j] = S[N2 + j]; sz[j] = S[2 * N2 + j];
  }
  __syncthreads();
  const int n = blockIdx.x * 256 + tid;
  const float* D = dxyz + (size_t)b * 3 * N1;
  const float px = D[n], py = D[N1 + n], pz = D[2 * N1 + n];
  float best0 = 3.4e38f, best1 = 3.4e38f, best2 = 3.4e38f;
  int j0 = 0, j1 = 0, j2 = 0;
  for (int j = 0; j < N2; ++j) {
    float dx = px - sx[j], dy = py - sy[j], dz = pz - sz[j];
    float d = fmaf(dx, dx, fmaf(dy, dy, dz * dz));
    if (d < best2) {
      if (d < best1) {
        if (d < best0) { best2 = best1; j2 = j1; best1 = best0; j1 = j0; best0 = d; j0 = j; }
        else           { best2 = best1; j2 = j1; best1 = d;     j1 = j; }
      } else           { best2 = d;     j2 = j; }
    }
  }
  float i0 = 1.f / fmaxf(best0, 1e-10f);
  float i1 = 1.f / fmaxf(best1, 1e-10f);
  float i2 = 1.f / fmaxf(best2, 1e-10f);
  float s = i0 + i1 + i2;
  size_t base = ((size_t)b * N1 + n) * 3;
  idx[base] = j0; idx[base + 1] = j1; idx[base + 2] = j2;
  wgt[base] = i0 / s; wgt[base + 1] = i1 / s; wgt[base + 2] = i2 / s;
}

// --------------------------------------------- interpolate + concat -> bf16 X0
// X0 layout: (B, 384, N1): ch 0..255 = interp(sparse_feature), 256..383 = dense
__global__ __launch_bounds__(256) void interp_concat_kernel(
    const float* __restrict__ sparse_feature, const float* __restrict__ dense_feature,
    const int* __restrict__ idx, const float* __restrict__ wgt,
    bf16_t* __restrict__ X0) {
  const int b = blockIdx.y;
  const int n = blockIdx.x * 256 + threadIdx.x;
  size_t base = ((size_t)b * N1 + n) * 3;
  const int j0 = idx[base], j1 = idx[base + 1], j2 = idx[base + 2];
  const float w0 = wgt[base], w1 = wgt[base + 1], w2 = wgt[base + 2];
  const float* F = sparse_feature + (size_t)b * C2 * N2;
  bf16_t* Xo = X0 + (size_t)b * CIN0 * N1;
  for (int c = 0; c < C2; ++c) {
    const float* Fr = F + (size_t)c * N2;          // 16KB row, L2-resident
    float v = fmaf(w0, Fr[j0], fmaf(w1, Fr[j1], w2 * Fr[j2]));
    Xo[(size_t)c * N1 + n] = (bf16_t)v;
  }
  const float* Dn = dense_feature + (size_t)b * C1 * N1 + n;
  for (int c = 0; c < C1; ++c)
    Xo[(size_t)(C2 + c) * N1 + n] = (bf16_t)Dn[(size_t)c * N1];
}

// ----------------------------------------------------------- WMMA bf16 GEMM
// Y[b, o, n] = sum_c W[o,c] * X[b,c,n].  Block tile 128(M) x 128(N), 8 waves
// arranged 4(M) x 2(N); wave tile 32 x 64 = 2x4 v_wmma_f32_16x16x32_bf16
// tiles; K-slab = 32.  A staged via async global->LDS when available.
__global__ __launch_bounds__(256) void wmma_gemm_kernel(
    const bf16_t* __restrict__ W, const bf16_t* __restrict__ X,
    float* __restrict__ Y, int Kdim) {
  __shared__ __align__(32) bf16_t As[128][32];   // weights, row-major [m][k]
  __shared__ __align__(32) bf16_t Bs[128][32];   // activations, transposed [n][k]

  const int tid = threadIdx.x;
  const int b = blockIdx.z;
  const int nBlock = blockIdx.x * 128;
  const int mBlock = blockIdx.y * 128;
  const bf16_t* Xb = X + (size_t)b * Kdim * N1;
  float* Yb = Y + (size_t)b * COUT * N1;

  const int lane = tid & 31, wave = tid >> 5;
  const int wy = wave >> 1;          // 0..3 -> M offset wy*32
  const int wx = wave & 1;           // 0..1 -> N offset wx*64
  const int lm = lane & 15;          // row/col within 16x16 tile
  const int lh = lane >> 4;          // K-half select

  v8f acc[2][4];
#pragma unroll
  for (int i = 0; i < 2; ++i)
#pragma unroll
    for (int j = 0; j < 4; ++j)
#pragma unroll
      for (int e = 0; e < 8; ++e) acc[i][j][e] = 0.f;

  // staging index assignments
  const int ar = tid >> 1;             // A row 0..127
  const int ah = (tid & 1) * 16;       // A k-half 0/16
  const int tk = tid >> 6;             // B k row 0..3
  const int tn = (tid & 63) * 2;       // B n col (2 per thread), 0..126

  for (int k0 = 0; k0 < Kdim; k0 += 32) {
    __syncthreads();
    // stage A: 128x32 bf16 (8KB) straight copy
    const bf16_t* wsrc = W + (size_t)(mBlock + ar) * Kdim + k0 + ah;
#ifdef USE_ASYNC_LDS
    __builtin_amdgcn_global_load_async_to_lds_b128(
        (gas_v4i*)wsrc, (las_v4i*)&As[ar][ah], 0, 0);
    __builtin_amdgcn_global_load_async_to_lds_b128(
        (gas_v4i*)(wsrc + 8), (las_v4i*)&As[ar][ah + 8], 0, 0);
#else
    *(uint4*)&As[ar][ah]     = *(const uint4*)wsrc;
    *(uint4*)&As[ar][ah + 8] = *(const uint4*)(wsrc + 8);
#endif
    // stage B transposed: read coalesced along n, write [n][k]
#pragma unroll
    for (int kk = 0; kk < 32; kk += 4) {
      const int k = kk + tk;
      union { unsigned u; bf16_t h[2]; } p;
      p.u = *(const unsigned*)(Xb + (size_t)(k0 + k) * N1 + nBlock + tn);
      Bs[tn][k]     = p.h[0];
      Bs[tn + 1][k] = p.h[1];
    }
    if (k0 + 32 < Kdim) {
      __builtin_prefetch(Xb + (size_t)(k0 + 32) * N1 + nBlock + tn, 0, 1);
      __builtin_prefetch(W + (size_t)(mBlock + ar) * Kdim + k0 + 32 + ah, 0, 1);
    }
#ifdef USE_ASYNC_LDS
    __builtin_amdgcn_s_wait_asynccnt(0);
#endif
    __syncthreads();

    // build fragments in contiguous k order, k = lh*16 + s, same for A and B (wave32)
    v16bf afrg[2], bfrg[4];
#pragma unroll
    for (int i = 0; i < 2; ++i) {
      const int m = wy * 32 + i * 16 + lm;
      v8bf alo = *(const v8bf*)&As[m][lh * 16];        // K = lh*16 .. +7
      v8bf ahi = *(const v8bf*)&As[m][lh * 16 + 8];    // K = lh*16+8 .. +15
      afrg[i] = __builtin_shufflevector(alo, ahi,
                    0, 1, 2, 3, 4, 5, 6, 7, 8, 9, 10, 11, 12, 13, 14, 15);
    }
#pragma unroll
    for (int j = 0; j < 4; ++j) {
      const int nn = wx * 64 + j * 16 + lm;
      bfrg[j] = *(const v16bf*)&Bs[nn][lh * 16];       // K = lh*16 .. +15
    }
#pragma unroll
    for (int i = 0; i < 2; ++i)
#pragma unroll
      for (int j = 0; j < 4; ++j)
        acc[i][j] = __builtin_amdgcn_wmma_f32_16x16x32_bf16(
            false, afrg[i], false, bfrg[j], (short)0, acc[i][j], false, false);
  }

  // epilogue: C/D layout — VGPR e, lane L => (m = e + 8*lh, n = lm) within tile
#pragma unroll
  for (int i = 0; i < 2; ++i)
#pragma unroll
    for (int j = 0; j < 4; ++j) {
      const int row0 = mBlock + wy * 32 + i * 16 + lh * 8;
      const int col  = nBlock + wx * 64 + j * 16 + lm;
#pragma unroll
      for (int e = 0; e < 8; ++e)
        Yb[(size_t)(row0 + e) * N1 + col] = acc[i][j][e];
    }
}

// ------------------------------------------------ per-channel BN statistics
__global__ __launch_bounds__(256) void stats_kernel(
    const float* __restrict__ Y, float* __restrict__ stats) {
  const int o = blockIdx.x;
  const int tid = threadIdx.x;
  float s = 0.f, s2 = 0.f;
  for (int e = tid; e < BATCH * N1; e += 256) {
    const int bb = e >> 14;          // N1 = 2^14
    const int n  = e & (N1 - 1);
    float v = Y[((size_t)(bb * COUT + o)) * N1 + n];
    s += v; s2 += v * v;
  }
  __shared__ float rs[256], rq[256];
  rs[tid] = s; rq[tid] = s2;
  __syncthreads();
  for (int off = 128; off > 0; off >>= 1) {
    if (tid < off) { rs[tid] += rs[tid + off]; rq[tid] += rq[tid + off]; }
    __syncthreads();
  }
  if (tid == 0) {
    const float inv_n = 1.f / (float)(BATCH * N1);
    float m = rs[0] * inv_n;
    float var = rq[0] * inv_n - m * m;
    stats[o] = m;
    stats[COUT + o] = rsqrtf(var + BN_EPS);
  }
}

// ------------------------------------------------------- BN + ReLU epilogues
__device__ __forceinline__ float bn_relu_val(
    float v, int c, const float* g, const float* bb, const float* stats) {
  float xh = (v - stats[c]) * stats[COUT + c];
  return fmaxf(fmaf(g[c], xh, bb[c]), 0.f);
}

__global__ __launch_bounds__(256) void bn_relu_bf16_kernel(
    const float* __restrict__ Y, const float* __restrict__ g,
    const float* __restrict__ bb, const float* __restrict__ stats,
    bf16_t* __restrict__ Xn) {
  const int total = BATCH * COUT * N1;
  int e = blockIdx.x * 256 + threadIdx.x;
  const int stride = gridDim.x * 256;
  for (; e < total; e += stride) {
    const int c = (e >> 14) & (COUT - 1);
    Xn[e] = (bf16_t)bn_relu_val(Y[e], c, g, bb, stats);
  }
}

__global__ __launch_bounds__(256) void bn_relu_f32_kernel(
    const float* __restrict__ Y, const float* __restrict__ g,
    const float* __restrict__ bb, const float* __restrict__ stats,
    float* __restrict__ out) {
  const int total = BATCH * COUT * N1;
  int e = blockIdx.x * 256 + threadIdx.x;
  const int stride = gridDim.x * 256;
  for (; e < total; e += stride) {
    const int c = (e >> 14) & (COUT - 1);
    out[e] = bn_relu_val(Y[e], c, g, bb, stats);
  }
}

// ---------------------------------------------------------------------------
extern "C" void kernel_launch(void* const* d_in, const int* in_sizes, int n_in,
                              void* d_out, int out_size, void* d_ws, size_t ws_size,
                              hipStream_t stream) {
  const float* dense_xyz      = (const float*)d_in[0];
  const float* sparse_xyz     = (const float*)d_in[1];
  const float* dense_feature  = (const float*)d_in[2];
  const float* sparse_feature = (const float*)d_in[3];
  const float* W0 = (const float*)d_in[4];
  const float* g0 = (const float*)d_in[5];
  const float* b0 = (const float*)d_in[6];
  const float* W1 = (const float*)d_in[7];
  const float* g1 = (const float*)d_in[8];
  const float* b1 = (const float*)d_in[9];
  const float* W2 = (const float*)d_in[10];
  const float* g2 = (const float*)d_in[11];
  const float* b2 = (const float*)d_in[12];

  char* ws = (char*)d_ws;
  size_t off = 0;
  auto walloc = [&](size_t bytes) -> void* {
    void* p = ws + off;
    off = (off + bytes + 255) & ~(size_t)255;
    return p;
  };
  bf16_t* Wb0     = (bf16_t*)walloc((size_t)COUT * CIN0 * 2);
  bf16_t* Wb1     = (bf16_t*)walloc((size_t)COUT * COUT * 2);
  bf16_t* Wb2     = (bf16_t*)walloc((size_t)COUT * COUT * 2);
  int*    knn_idx = (int*)   walloc((size_t)BATCH * N1 * 3 * 4);
  float*  knn_w   = (float*) walloc((size_t)BATCH * N1 * 3 * 4);
  float*  stats   = (float*) walloc((size_t)2 * COUT * 4);
  bf16_t* X0      = (bf16_t*)walloc((size_t)BATCH * CIN0 * N1 * 2);
  bf16_t* X1      = (bf16_t*)walloc((size_t)BATCH * COUT * N1 * 2);
  bf16_t* X2      = (bf16_t*)walloc((size_t)BATCH * COUT * N1 * 2);
  float*  Yb      = (float*) walloc((size_t)BATCH * COUT * N1 * 4);
  (void)ws_size; (void)in_sizes; (void)n_in; (void)out_size;

  // weight conversion
  f32_to_bf16_kernel<<<96, 256, 0, stream>>>(W0, Wb0, COUT * CIN0);
  f32_to_bf16_kernel<<<64, 256, 0, stream>>>(W1, Wb1, COUT * COUT);
  f32_to_bf16_kernel<<<64, 256, 0, stream>>>(W2, Wb2, COUT * COUT);

  // 3-NN + interpolation + concat
  knn_kernel<<<dim3(N1 / 256, BATCH), 256, 0, stream>>>(
      dense_xyz, sparse_xyz, knn_idx, knn_w);
  interp_concat_kernel<<<dim3(N1 / 256, BATCH), 256, 0, stream>>>(
      sparse_feature, dense_feature, knn_idx, knn_w, X0);

  const dim3 gemm_grid(N1 / 128, COUT / 128, BATCH);

  // layer 0: (256 x 384) @ (384 x N)
  wmma_gemm_kernel<<<gemm_grid, 256, 0, stream>>>(Wb0, X0, Yb, CIN0);
  stats_kernel<<<COUT, 256, 0, stream>>>(Yb, stats);
  bn_relu_bf16_kernel<<<4096, 256, 0, stream>>>(Yb, g0, b0, stats, X1);

  // layer 1: (256 x 256) @ (256 x N)
  wmma_gemm_kernel<<<gemm_grid, 256, 0, stream>>>(Wb1, X1, Yb, COUT);
  stats_kernel<<<COUT, 256, 0, stream>>>(Yb, stats);
  bn_relu_bf16_kernel<<<4096, 256, 0, stream>>>(Yb, g1, b1, stats, X2);

  // layer 2: (256 x 256) @ (256 x N), f32 output
  wmma_gemm_kernel<<<gemm_grid, 256, 0, stream>>>(Wb2, X2, Yb, COUT);
  stats_kernel<<<COUT, 256, 0, stream>>>(Yb, stats);
  bn_relu_f32_kernel<<<4096, 256, 0, stream>>>(Yb, g2, b2, stats, (float*)d_out);
}
